// AttentionHead_57767310131667
// MI455X (gfx1250) — matrix operand order repaired and checked by reading, not compiled, not run on hardware
//
#include <hip/hip_runtime.h>
#include <hip/hip_bf16.h>

// ---------------------------------------------------------------------------
// Causal attention head, column-axis softmax (softmax over queries), MI455X.
// wave32 + v_wmma_f32_16x16x32_bf16 for all GEMM stages.
// q/k/v cached in bf16 workspace (6 MB -> L2 resident); score tiles recomputed
// instead of materializing the 128 MB `wei` matrix.
// Round 3: software pipelining -- distinct registers for all in-flight
// fragments so global_load_b128 clauses run ahead of the WMMAs that consume
// them (removes the s_wait_loadcnt 0x0 stall before every WMMA).
// ---------------------------------------------------------------------------

typedef __attribute__((ext_vector_type(16))) __bf16 v16bf;
typedef __attribute__((ext_vector_type(8)))  __bf16 v8bf;
typedef __attribute__((ext_vector_type(8)))  float  v8f;
typedef __attribute__((ext_vector_type(4)))  float  v4f;

constexpr int BATCH = 8;
constexpr int TSEQ  = 2048;
constexpr int EMB   = 1024;
constexpr int HDIM  = 64;
constexpr int KCH   = EMB / 32;       // 32 K-chunks of 32
constexpr float QSCALE  = 0.03125f;   // 1024^-0.5, folded into packed Wq
constexpr float NEG_BIG = -1.0e30f;   // finite -inf stand-in (avoids inf-inf NaN)

#define WMMA_BF16(A, B, C) \
  __builtin_amdgcn_wmma_f32_16x16x32_bf16(false, (A), false, (B), (short)0, (C), false, false)

#define SHUF16(LO, HI) __builtin_shufflevector((LO), (HI), 0, 1, 2, 3, 4, 5, 6, 7, \
                                               8, 9, 10, 11, 12, 13, 14, 15)

__device__ inline v8f vzero8() {
  v8f z;
#pragma unroll
  for (int e = 0; e < 8; ++e) z[e] = 0.0f;
  return z;
}

// ---------------------------------------------------------------------------
// Kernel 0: pack Wq/Wk/Wv (fp32 [E][H]) into bf16 B-fragment order:
//   Wpk[mat][kc][nt][lane][j] = W[(kc*32 + (lane>>4)*8 + (j&7) + (j>>3)*16)*H + nt*16 + (lane&15)]
// One wave per (kc, nt, mat); 1/sqrt(E) folded into Wq.
// ---------------------------------------------------------------------------
__global__ __launch_bounds__(32) void wpack_kernel(
    const float* __restrict__ Wq, const float* __restrict__ Wk, const float* __restrict__ Wv,
    __bf16* __restrict__ Wpk) {
  const int lane = threadIdx.x;
  const int n  = lane & 15;
  const int hi = lane >> 4;
  const int kc  = blockIdx.x;          // 0..KCH-1
  const int nt  = blockIdx.y;          // 0..3
  const int mat = blockIdx.z;          // 0:q 1:k 2:v
  const float* Ws[3] = {Wq, Wk, Wv};
  const float scl = (mat == 0) ? QSCALE : 1.0f;
  const float* src = Ws[mat] + (size_t)(kc * 32 + hi * 8) * HDIM + nt * 16 + n;
  __bf16* dst = Wpk + ((((size_t)mat * KCH + kc) * 4 + nt) * 32 + lane) * 16;
  v8bf lo, hi8;
#pragma unroll
  for (int j = 0; j < 8; ++j) {
    lo[j]  = (__bf16)(src[j * HDIM] * scl);
    hi8[j] = (__bf16)(src[(j + 16) * HDIM] * scl);
  }
  *(v8bf*)(dst)     = lo;
  *(v8bf*)(dst + 8) = hi8;
}

// ---------------------------------------------------------------------------
// Kernel 1: fused q/k/v projection; one wave per 16-row tile of x.
// All 12 weight B-frags of a k-chunk preloaded into distinct registers
// (one long load clause) before the 12 WMMAs -> loads overlap matrix ops.
// q,k stored row-major [r][H]; v stored transposed vT[b][h][t].
// ---------------------------------------------------------------------------
__global__ __launch_bounds__(32) void qkv_proj_kernel(
    const float* __restrict__ x, const __bf16* __restrict__ Wpk,
    __bf16* __restrict__ qB, __bf16* __restrict__ kB, __bf16* __restrict__ vT) {
  const int lane = threadIdx.x;
  const int n  = lane & 15;
  const int hi = lane >> 4;
  const int rowTile = blockIdx.x;
  const int arow = rowTile * 16 + n;

  v8f acc[3][4];
#pragma unroll
  for (int m = 0; m < 3; ++m)
#pragma unroll
    for (int nt = 0; nt < 4; ++nt) acc[m][nt] = vzero8();

  const float* xrow = x + (size_t)arow * EMB;
  const __bf16* wlane = Wpk + (size_t)lane * 16;

  for (int kc = 0; kc < KCH; ++kc) {
    const int kk = kc * 32;
    // Issue x loads first
    v4f x0 = *(const v4f*)(xrow + kk + hi * 8);
    v4f x1 = *(const v4f*)(xrow + kk + hi * 8 + 4);
    v4f x2 = *(const v4f*)(xrow + kk + 16 + hi * 8);
    v4f x3 = *(const v4f*)(xrow + kk + 16 + hi * 8 + 4);
    // Preload all 12 weight B-frags into distinct registers (long clause)
    v16bf b[12];
#pragma unroll
    for (int f = 0; f < 12; ++f) {
      const int m  = f >> 2;
      const int nt = f & 3;
      b[f] = *(const v16bf*)(wlane + ((((size_t)m * KCH + kc) * 4 + nt) * 32) * 16);
    }
    // Convert x to bf16 A-frag while weight loads are in flight
    v16bf a;
#pragma unroll
    for (int j = 0; j < 4; ++j) {
      a[j]      = (__bf16)x0[j];
      a[j + 4]  = (__bf16)x1[j];
      a[j + 8]  = (__bf16)x2[j];
      a[j + 12] = (__bf16)x3[j];
    }
    // 12 back-to-back WMMAs; waits can be incremental since b[] are distinct
#pragma unroll
    for (int f = 0; f < 12; ++f)
      acc[f >> 2][f & 3] = WMMA_BF16(a, b[f], acc[f >> 2][f & 3]);
  }

  const int bidx  = (rowTile * 16) / TSEQ;
  const int tbase = rowTile * 16 - bidx * TSEQ;
#pragma unroll
  for (int nt = 0; nt < 4; ++nt) {
#pragma unroll
    for (int i = 0; i < 8; ++i) {
      const size_t r = (size_t)rowTile * 16 + i + hi * 8;
      qB[r * HDIM + nt * 16 + n] = (__bf16)acc[0][nt][i];
      kB[r * HDIM + nt * 16 + n] = (__bf16)acc[1][nt][i];
    }
    // v transposed: h = nt*16+n fixed per lane, t contiguous -> one 16B store
    v8bf vv;
#pragma unroll
    for (int i = 0; i < 8; ++i) vv[i] = (__bf16)acc[2][nt][i];
    *(v8bf*)(vT + ((size_t)bidx * HDIM + nt * 16 + n) * TSEQ + tbase + hi * 8) = vv;
  }
}

// ---------------------------------------------------------------------------
// Kernel 2: per-key-column online max / sum(exp) over queries t >= s.
// One wave owns (batch, 16-key tile). Software-pipelined: next query tile's
// A-frag loads are issued before the current tile's WMMAs + stats VALU work.
// ---------------------------------------------------------------------------
__global__ __launch_bounds__(32) void colstats_kernel(
    const __bf16* __restrict__ qB, const __bf16* __restrict__ kB,
    float* __restrict__ colmax, float* __restrict__ colsum) {
  const int lane = threadIdx.x;
  const int n  = lane & 15;
  const int hi = lane >> 4;
  const int s0 = blockIdx.x * 16;
  const size_t base = (size_t)blockIdx.y * TSEQ;

  // Loop-invariant k^T B-frags: contiguous 32B per lane
  const __bf16* kp = kB + (base + s0 + n) * HDIM + hi * 16;
  const v16bf bk0 = *(const v16bf*)kp;
  const v16bf bk1 = *(const v16bf*)(kp + 32);

  // Prologue: load A-frags for first tile
  const __bf16* qp = qB + (base + s0 + n) * HDIM + hi * 16;
  v8bf q0 = *(const v8bf*)(qp);
  v8bf q1 = *(const v8bf*)(qp + 8);
  v8bf q2 = *(const v8bf*)(qp + 32);
  v8bf q3 = *(const v8bf*)(qp + 40);

  float M = NEG_BIG, Ssum = 0.0f;
  for (int t0 = s0; t0 < TSEQ; t0 += 16) {
    v16bf a0 = SHUF16(q0, q1);
    v16bf a1 = SHUF16(q2, q3);
    // Prefetch next tile's A-frags (uniform branch; overlaps WMMA + stats)
    if (t0 + 16 < TSEQ) {
      const __bf16* qn = qB + (base + t0 + 16 + n) * HDIM + hi * 16;
      q0 = *(const v8bf*)(qn);
      q1 = *(const v8bf*)(qn + 8);
      q2 = *(const v8bf*)(qn + 32);
      q3 = *(const v8bf*)(qn + 40);
    }
    v8f c = vzero8();
    c = WMMA_BF16(a0, bk0, c);
    c = WMMA_BF16(a1, bk1, c);

    // lane column s = s0+n; rows t = t0 + i + hi*8; causal: valid iff t >= s
    float vals[8], bm = NEG_BIG;
#pragma unroll
    for (int i = 0; i < 8; ++i) {
      const int t = t0 + i + hi * 8;
      vals[i] = (t >= s0 + n) ? c[i] : NEG_BIG;
      bm = fmaxf(bm, vals[i]);
    }
    bm = fmaxf(bm, __shfl_xor(bm, 16, 32));
    const float newM = fmaxf(M, bm);      // finite from first (diagonal) tile on
    float ls = 0.0f;
#pragma unroll
    for (int i = 0; i < 8; ++i) ls += __expf(vals[i] - newM);  // masked -> 0
    ls += __shfl_xor(ls, 16, 32);
    Ssum = Ssum * __expf(M - newM) + ls;
    M = newM;
  }
  if (hi == 0) {
    colmax[base + s0 + n] = M;
    colsum[base + s0 + n] = Ssum;
  }
}

// ---------------------------------------------------------------------------
// Kernel 3: out[t,:] = sum_s p[t,s]*v[s,:], p[t,s] = exp(S-colmax[s])/colsum[s].
// S computed TRANSPOSED (A = k rows-s, B = q^T): the f32 C-fragment of S'
// is bit-for-bit the bf16 A-fragment layout for the p@v WMMA.
// Software-pipelined: next s-chunk's k A-frags + v B-frags prefetched before
// the current chunk's 8 WMMAs + exp/normalize.
// ---------------------------------------------------------------------------
__global__ __launch_bounds__(32) void attn_out_kernel(
    const __bf16* __restrict__ qB, const __bf16* __restrict__ kB, const __bf16* __restrict__ vT,
    const float* __restrict__ colmax, const float* __restrict__ colsum,
    float* __restrict__ out) {
  const int lane = threadIdx.x;
  const int n  = lane & 15;
  const int hi = lane >> 4;
  const int t0 = blockIdx.x * 16;
  const int bidx = blockIdx.y;
  const size_t base = (size_t)bidx * TSEQ;
  const int t = t0 + n;                   // S' column / output row for this lane

  // Loop-invariant q^T B-frags
  const __bf16* qp = qB + (base + t) * HDIM + hi * 8;
  const v16bf bq0 = SHUF16(*(const v8bf*)qp, *(const v8bf*)(qp + 16));
  const v16bf bq1 = SHUF16(*(const v8bf*)(qp + 32), *(const v8bf*)(qp + 48));

  v8f acc[4];
#pragma unroll
  for (int ht = 0; ht < 4; ++ht) acc[ht] = vzero8();

  // Prologue: prefetch first s-chunk's k A-frags (2 halves x 4) and v B-frags (4)
  v8bf kf[8];
  v16bf bv[4];
  {
    const __bf16* kp0 = kB + (base + n) * HDIM + hi * 8;        // ss = 0
    const __bf16* kp1 = kB + (base + 16 + n) * HDIM + hi * 8;   // ss = 16
#pragma unroll
    for (int j = 0; j < 4; ++j) {
      kf[j]     = *(const v8bf*)(kp0 + 16 * j);
      kf[4 + j] = *(const v8bf*)(kp1 + 16 * j);
    }
#pragma unroll
    for (int ht = 0; ht < 4; ++ht)
      bv[ht] = SHUF16(*(const v8bf*)(vT + ((size_t)bidx * HDIM + ht * 16 + n) * TSEQ + hi * 8),
                      *(const v8bf*)(vT + ((size_t)bidx * HDIM + ht * 16 + n) * TSEQ + 16 + hi * 8));
  }

  for (int s0 = 0; s0 <= t0; s0 += 32) {  // covers all keys s <= t0+15
    // Current chunk's fragments
    v16bf a00 = SHUF16(kf[0], kf[1]);
    v16bf a01 = SHUF16(kf[2], kf[3]);
    v16bf a10 = SHUF16(kf[4], kf[5]);
    v16bf a11 = SHUF16(kf[6], kf[7]);
    v16bf cv[4];
#pragma unroll
    for (int ht = 0; ht < 4; ++ht) cv[ht] = bv[ht];

    // Prefetch next chunk (uniform branch; hides under 8 WMMAs + exp work)
    if (s0 + 32 <= t0) {
      const int sn = s0 + 32;
      const __bf16* kp0 = kB + (base + sn + n) * HDIM + hi * 8;
      const __bf16* kp1 = kB + (base + sn + 16 + n) * HDIM + hi * 8;
#pragma unroll
      for (int j = 0; j < 4; ++j) {
        kf[j]     = *(const v8bf*)(kp0 + 16 * j);
        kf[4 + j] = *(const v8bf*)(kp1 + 16 * j);
      }
#pragma unroll
      for (int ht = 0; ht < 4; ++ht)
        bv[ht] = SHUF16(*(const v8bf*)(vT + ((size_t)bidx * HDIM + ht * 16 + n) * TSEQ
                                       + sn + hi * 8),
                        *(const v8bf*)(vT + ((size_t)bidx * HDIM + ht * 16 + n) * TSEQ
                                       + sn + 16 + hi * 8));
    }

    // S' tiles for both 16-row halves of the 32-key chunk
    v8f pc[2];
#pragma unroll
    for (int half = 0; half < 2; ++half) {
      const int ss = s0 + 16 * half;
      v8f c = vzero8();
      c = WMMA_BF16(half ? a10 : a00, bq0, c);
      c = WMMA_BF16(half ? a11 : a01, bq1, c);
      // rows s = ss + i + hi*8, col t; per-key stats as 32B vector loads
      v8f cm = *(const v8f*)(colmax + base + ss + hi * 8);
      v8f cs = *(const v8f*)(colsum + base + ss + hi * 8);
#pragma unroll
      for (int i = 0; i < 8; ++i) {
        const int s = ss + i + hi * 8;
        pc[half][i] = (s <= t) ? (__expf(c[i] - cm[i]) / cs[i]) : 0.0f;
      }
    }
    // p A-frag: direct C'->A register remap (no cross-lane movement)
    v16bf ap;
#pragma unroll
    for (int j = 0; j < 8; ++j) {
      ap[j]     = (__bf16)pc[0][j];
      ap[j + 8] = (__bf16)pc[1][j];
    }
    // Accumulate p @ v
#pragma unroll
    for (int ht = 0; ht < 4; ++ht)
      acc[ht] = WMMA_BF16(ap, cv[ht], acc[ht]);
  }

  // D-frag store: lane col n -> h = ht*16+n, VGPR i -> row t0 + i + hi*8
#pragma unroll
  for (int ht = 0; ht < 4; ++ht)
#pragma unroll
    for (int i = 0; i < 8; ++i)
      out[(base + t0 + i + hi * 8) * HDIM + ht * 16 + n] = acc[ht][i];
}

// ---------------------------------------------------------------------------
extern "C" void kernel_launch(void* const* d_in, const int* in_sizes, int n_in,
                              void* d_out, int out_size, void* d_ws, size_t ws_size,
                              hipStream_t stream) {
  const float* x  = (const float*)d_in[0];
  const float* Wk = (const float*)d_in[1];
  const float* Wq = (const float*)d_in[2];
  const float* Wv = (const float*)d_in[3];
  float* out = (float*)d_out;

  // Workspace: qB/kB/vT bf16 (3 x 2 MB) + colmax/colsum f32 (2 x 64 KB) + Wpk (384 KB)
  char* ws = (char*)d_ws;
  const size_t qkvBytes = (size_t)BATCH * TSEQ * HDIM * sizeof(__bf16);
  __bf16* qB = (__bf16*)(ws);
  __bf16* kB = (__bf16*)(ws + qkvBytes);
  __bf16* vT = (__bf16*)(ws + 2 * qkvBytes);
  float* colmax = (float*)(ws + 3 * qkvBytes);
  float* colsum = colmax + (size_t)BATCH * TSEQ;
  __bf16* Wpk = (__bf16*)(ws + 3 * qkvBytes + 2 * (size_t)BATCH * TSEQ * sizeof(float));

  dim3 gw(KCH, 4, 3);
  wpack_kernel<<<gw, 32, 0, stream>>>(Wq, Wk, Wv, Wpk);

  const int rowTiles = BATCH * TSEQ / 16;            // 1024 waves
  qkv_proj_kernel<<<rowTiles, 32, 0, stream>>>(x, Wpk, qB, kB, vT);

  dim3 g2(TSEQ / 16, BATCH);                          // 1024 waves each
  colstats_kernel<<<g2, 32, 0, stream>>>(qB, kB, colmax, colsum);
  attn_out_kernel<<<g2, 32, 0, stream>>>(qB, kB, vT, colmax, colsum, out);
}
